// VectorQuantizerEMA_65000035058428
// MI455X (gfx1250) — matrix operand order repaired and checked by reading, not compiled, not run on hardware
//
#include <hip/hip_runtime.h>
#include <cstdint>
#include <cstddef>

// ---------------------------------------------------------------------------
// VQ-VAE EMA vector quantizer, MI455X (gfx1250, wave32, WMMA).
// Distance GEMM runs on v_wmma_f32_16x16x32_bf16 with bf16x3 error
// compensation (hi*hi + lo*hi + hi*lo, f32 accumulate) fused with argmin.
// Round 3: B-fragment LDS loads software-pipelined one step ahead of the
// consuming WMMAs (plus the round-2 double-buffered LDS global pipeline and
// dual accumulator chains).
// ---------------------------------------------------------------------------

#define NUM_EMB   8192
#define EMB_DIM   256
#define BATCH     16
#define HW        1024            // 32*32
#define NTOK      16384           // BATCH*HW
#define DECAY     0.99f
#define OMD       0.01f           // 1-DECAY
#define COMMIT    0.25f
#define EPS_F     1e-5f
#define NCHUNK    (NUM_EMB / 16)  // 512 16-code chunks

typedef __bf16 bf16_t;
typedef __attribute__((ext_vector_type(8)))  __bf16 v8bf;
typedef __attribute__((ext_vector_type(16))) __bf16 v16bf;
typedef __attribute__((ext_vector_type(8)))  float  v8f;

// d_out layout (floats, concatenated reference outputs in return order)
static constexpr size_t OFF_LOSS = 0;
static constexpr size_t OFF_QST  = 1;
static constexpr size_t OFF_PERP = OFF_QST + (size_t)BATCH * EMB_DIM * HW;        // 4194305
static constexpr size_t OFF_EMB  = OFF_PERP + 1;                                   // 4194306
static constexpr size_t OFF_CS   = OFF_EMB + (size_t)NUM_EMB * EMB_DIM;            // 6291458
static constexpr size_t OFF_EMAW = OFF_CS + NUM_EMB;                               // 6299650
static constexpr size_t OFF_IDX  = OFF_EMAW + (size_t)NUM_EMB * EMB_DIM;           // 8396802

static __device__ __forceinline__ v8f wmma_bf16(v16bf a, v16bf b, v8f c) {
  // (neg_a, A, neg_b, B, c_mod, C, reuse_a, reuse_b)
  return __builtin_amdgcn_wmma_f32_16x16x32_bf16(false, a, false, b, (short)0, c,
                                                 false, false);
}

static __device__ __forceinline__ void split_bf16(float f, bf16_t& h, bf16_t& l) {
  h = (bf16_t)f;
  l = (bf16_t)(f - (float)h);
}

// ---------------------------------------------------------------------------
// 1) Codebook prep: embedding[K,D] f32 -> ehi/elo bf16, e2[K] = ||e_k||^2
// ---------------------------------------------------------------------------
__global__ __launch_bounds__(256) void k_prep_codebook(
    const float* __restrict__ emb, bf16_t* __restrict__ ehi,
    bf16_t* __restrict__ elo, float* __restrict__ e2) {
  const int k = blockIdx.x, d = threadIdx.x;
  const size_t i = (size_t)k * EMB_DIM + d;
  float e = emb[i];
  bf16_t h, l;
  split_bf16(e, h, l);
  ehi[i] = h;
  elo[i] = l;
  float s = e * e;
  for (int off = 16; off; off >>= 1) s += __shfl_down(s, off, 32);
  __shared__ float red[8];
  if ((threadIdx.x & 31) == 0) red[threadIdx.x >> 5] = s;
  __syncthreads();
  if (threadIdx.x == 0) {
    float t = 0.f;
#pragma unroll
    for (int w = 0; w < 8; ++w) t += red[w];
    e2[k] = t;
  }
}

// ---------------------------------------------------------------------------
// 2) X prep: NCHW x -> token-major [NTOK, D] bf16 hi/lo (LDS tile transpose)
//    grid (HW/32, D/32, BATCH), block 256
// ---------------------------------------------------------------------------
__global__ __launch_bounds__(256) void k_prep_x(
    const float* __restrict__ x, bf16_t* __restrict__ xhi,
    bf16_t* __restrict__ xlo) {
  __shared__ float tile[32][33];
  const int hw0 = blockIdx.x * 32, d0 = blockIdx.y * 32, b = blockIdx.z;
  const int tr = threadIdx.x >> 5, tc = threadIdx.x & 31;
#pragma unroll
  for (int i = 0; i < 4; ++i) {
    const int dd = tr + i * 8;
    tile[dd][tc] = x[(size_t)b * EMB_DIM * HW + (size_t)(d0 + dd) * HW + hw0 + tc];
  }
  __syncthreads();
#pragma unroll
  for (int i = 0; i < 4; ++i) {
    const int hh = tr + i * 8;
    const float f = tile[tc][hh];
    bf16_t h, l;
    split_bf16(f, h, l);
    const size_t n = (size_t)b * HW + hw0 + hh;
    xhi[n * EMB_DIM + d0 + tc] = h;
    xlo[n * EMB_DIM + d0 + tc] = l;
  }
}

// ---------------------------------------------------------------------------
// 3) Fused distance GEMM + argmin (double-buffered software pipeline).
//    Block = 256 threads = 8 waves; wave w owns rows [blk*128 + w*16, +16).
//    Per 16-code chunk per wave: 24 x v_wmma_f32_16x16x32_bf16 split across
//    two independent accumulator chains; B fragments prefetched one step
//    ahead from LDS. Global loads for chunk kt+1 in flight during the
//    matrix block; LDS ping-pong, one barrier per iteration.
//    Relative distance: ||e||^2 - 2 * <x, e>  (||x||^2 is row-constant).
// ---------------------------------------------------------------------------
__global__ __launch_bounds__(256) void k_argmin(
    const bf16_t* __restrict__ xhi, const bf16_t* __restrict__ xlo,
    const bf16_t* __restrict__ ehi, const bf16_t* __restrict__ elo,
    const float* __restrict__ e2, int* __restrict__ idxI,
    float* __restrict__ idxF) {
  __shared__ __align__(32) bf16_t lHi[2][16 * EMB_DIM];   // 2 x 8 KB
  __shared__ __align__(32) bf16_t lLo[2][16 * EMB_DIM];   // 2 x 8 KB

  const int wave = threadIdx.x >> 5;
  const int lane = threadIdx.x & 31;
  const int rowBase = blockIdx.x * 128 + wave * 16;
  const int arow  = rowBase + (lane & 15);
  const int kboff = ((lane >> 4) & 1) * 16;  // contiguous k order, same as B: lanes 16-31 hold K=16-31

  // A fragments: 16 rows x 256 dims, hi+lo, resident in VGPRs (128 VGPRs).
  v16bf aHi[8], aLo[8];
#pragma unroll
  for (int c = 0; c < 8; ++c) {
    const bf16_t* ph = xhi + (size_t)arow * EMB_DIM + c * 32 + kboff;
    const bf16_t* pl = xlo + (size_t)arow * EMB_DIM + c * 32 + kboff;
    v8bf h0 = *(const v8bf*)ph;
    v8bf h1 = *(const v8bf*)(ph + 8);
    v8bf l0 = *(const v8bf*)pl;
    v8bf l1 = *(const v8bf*)(pl + 8);
    aHi[c] = __builtin_shufflevector(h0, h1, 0, 1, 2, 3, 4, 5, 6, 7,
                                     8, 9, 10, 11, 12, 13, 14, 15);
    aLo[c] = __builtin_shufflevector(l0, l1, 0, 1, 2, 3, 4, 5, 6, 7,
                                     8, 9, 10, 11, 12, 13, 14, 15);
  }

  float bestVal[8];
  int   bestIdx[8];
#pragma unroll
  for (int r = 0; r < 8; ++r) { bestVal[r] = 3.402823466e38f; bestIdx[r] = 0; }

  const int bcol = lane & 15;               // B/C column this lane serves
  const int koff = (lane >= 16) ? 16 : 0;   // contiguous k order, same as A: lanes 16-31 hold K=16-31

  const uint4* gH = (const uint4*)ehi;      // 512 uint4 per 16-code chunk
  const uint4* gL = (const uint4*)elo;
  const int t0 = threadIdx.x, t1 = threadIdx.x + 256;

  uint4 rH0, rH1, rL0, rL1;
  // prologue: stage chunk 0 into buffer 0
  {
    rH0 = gH[t0]; rH1 = gH[t1]; rL0 = gL[t0]; rL1 = gL[t1];
    uint4* dH = (uint4*)lHi[0];
    uint4* dL = (uint4*)lLo[0];
    dH[t0] = rH0; dH[t1] = rH1; dL[t0] = rL0; dL[t1] = rL1;
  }

  for (int kt = 0; kt < NCHUNK; ++kt) {
    __syncthreads();                        // buffer kt&1 ready; (kt+1)&1 free
    const int buf = kt & 1;

    // global loads for chunk kt+1 in flight during the WMMA block
    const bool more = (kt + 1) < NCHUNK;
    if (more) {
      const size_t base = (size_t)(kt + 1) * 512;
      rH0 = gH[base + t0]; rH1 = gH[base + t1];
      rL0 = gL[base + t0]; rL1 = gL[base + t1];
    }
    const int   n  = kt * 16 + bcol;
    const float en = e2[n];                 // issued before the matrix block

    v8f acc0 = {0.f, 0.f, 0.f, 0.f, 0.f, 0.f, 0.f, 0.f};
    v8f acc1 = {0.f, 0.f, 0.f, 0.f, 0.f, 0.f, 0.f, 0.f};
    const bf16_t* bH = lHi[buf] + bcol * EMB_DIM + koff;
    const bf16_t* bL = lLo[buf] + bcol * EMB_DIM + koff;

    // B fragments prefetched one c-step ahead of the consuming WMMAs
    v16bf bh = *(const v16bf*)(bH);
    v16bf bl = *(const v16bf*)(bL);
#pragma unroll
    for (int c = 0; c < 8; ++c) {
      v16bf bhN, blN;
      if (c < 7) {
        bhN = *(const v16bf*)(bH + (c + 1) * 32);
        blN = *(const v16bf*)(bL + (c + 1) * 32);
      }
      acc0 = wmma_bf16(aHi[c], bh, acc0);   // hi*hi chain
      acc1 = wmma_bf16(aLo[c], bh, acc1);   // cross-term chain
      acc1 = wmma_bf16(aHi[c], bl, acc1);
      if (c < 7) { bh = bhN; bl = blN; }
    }

    // write staged chunk kt+1 into the alternate buffer (reads of it ended
    // before the barrier at the top of this iteration)
    if (more) {
      uint4* dH = (uint4*)lHi[buf ^ 1];
      uint4* dL = (uint4*)lLo[buf ^ 1];
      dH[t0] = rH0; dH[t1] = rH1; dL[t0] = rL0; dL[t1] = rL1;
    }

#pragma unroll
    for (int r = 0; r < 8; ++r) {
      const float dval = en - 2.0f * (acc0[r] + acc1[r]);
      if (dval < bestVal[r]) { bestVal[r] = dval; bestIdx[r] = n; }
    }
  }

  // argmin across the 16-lane column group (prefer smaller index on ties)
#pragma unroll
  for (int m = 8; m >= 1; m >>= 1) {
#pragma unroll
    for (int r = 0; r < 8; ++r) {
      const float ov = __shfl_xor(bestVal[r], m, 16);
      const int   oi = __shfl_xor(bestIdx[r], m, 16);
      if (ov < bestVal[r] || (ov == bestVal[r] && oi < bestIdx[r])) {
        bestVal[r] = ov;
        bestIdx[r] = oi;
      }
    }
  }

  if ((lane & 15) == 0) {
    const int mb = (lane >= 16) ? 8 : 0;   // C layout: lanes 16-31 are M=8..15
#pragma unroll
    for (int r = 0; r < 8; ++r) {
      const int row = rowBase + mb + r;
      idxI[row] = bestIdx[r];
      idxF[row] = (float)bestIdx[r];
    }
  }
}

// ---------------------------------------------------------------------------
// 4) counts[k] += 1 per token
// ---------------------------------------------------------------------------
__global__ __launch_bounds__(256) void k_counts(const int* __restrict__ idxI,
                                                float* __restrict__ counts) {
  const int n = blockIdx.x * 256 + threadIdx.x;
  atomicAdd(&counts[idxI[n]], 1.0f);
}

// ---------------------------------------------------------------------------
// 5) q_st (== quantized), dw scatter, SSE for commitment loss.
//    One thread per NCHW element; x read + q_st write fully coalesced.
// ---------------------------------------------------------------------------
__global__ __launch_bounds__(256) void k_scatter(
    const float* __restrict__ x, const float* __restrict__ emb,
    const int* __restrict__ idxI, float* __restrict__ qst,
    float* __restrict__ dw, float* __restrict__ sse) {
  const int t  = blockIdx.x * 256 + threadIdx.x;        // NCHW linear index
  const int hw = t & (HW - 1);
  const int d  = (t >> 10) & (EMB_DIM - 1);
  const int b  = t >> 18;
  const int n  = b * HW + hw;
  const int k  = idxI[n];
  const float xv = x[t];
  const float q  = emb[(size_t)k * EMB_DIM + d];
  qst[t] = q;                                           // straight-through value
  atomicAdd(&dw[(size_t)k * EMB_DIM + d], xv);          // segment_sum(flat)
  const float diff = q - xv;
  float v = diff * diff;
  for (int off = 16; off; off >>= 1) v += __shfl_down(v, off, 32);
  __shared__ float red[8];
  if ((threadIdx.x & 31) == 0) red[threadIdx.x >> 5] = v;
  __syncthreads();
  if (threadIdx.x == 0) {
    float s = 0.f;
#pragma unroll
    for (int w = 0; w < 8; ++w) s += red[w];
    atomicAdd(sse, s);
  }
}

// ---------------------------------------------------------------------------
// 6) raw EMA cluster sizes, total n, entropy accumulator for perplexity
// ---------------------------------------------------------------------------
__global__ __launch_bounds__(256) void k_cs_ent(
    const float* __restrict__ counts, const float* __restrict__ emaCs,
    float* __restrict__ rawCs, float* __restrict__ nTot,
    float* __restrict__ ent) {
  const int k = blockIdx.x * 256 + threadIdx.x;
  const float c   = counts[k];
  const float raw = DECAY * emaCs[k] + OMD * c;
  rawCs[k] = raw;
  const float p = c / (float)NTOK;
  float rs = raw;
  float es = -p * logf(p + 1e-10f);
  for (int off = 16; off; off >>= 1) {
    rs += __shfl_down(rs, off, 32);
    es += __shfl_down(es, off, 32);
  }
  __shared__ float r1[8], r2[8];
  if ((threadIdx.x & 31) == 0) { r1[threadIdx.x >> 5] = rs; r2[threadIdx.x >> 5] = es; }
  __syncthreads();
  if (threadIdx.x == 0) {
    float a = 0.f, bsum = 0.f;
#pragma unroll
    for (int w = 0; w < 8; ++w) { a += r1[w]; bsum += r2[w]; }
    atomicAdd(nTot, a);
    atomicAdd(ent, bsum);
  }
}

// ---------------------------------------------------------------------------
// 7) Laplace-smoothed new_cs, loss and perplexity scalars
// ---------------------------------------------------------------------------
__global__ __launch_bounds__(256) void k_finalize(
    const float* __restrict__ rawCs, const float* __restrict__ nTot,
    const float* __restrict__ sse, const float* __restrict__ ent,
    float* __restrict__ out) {
  const int k = blockIdx.x * 256 + threadIdx.x;
  const float n = *nTot;
  out[OFF_CS + k] = (rawCs[k] + EPS_F) / (n + (float)NUM_EMB * EPS_F) * n;
  if (k == 0) {
    out[OFF_LOSS] = COMMIT * (*sse) / (float)((size_t)NTOK * EMB_DIM);
    out[OFF_PERP] = expf(*ent);
  }
}

// ---------------------------------------------------------------------------
// 8) new_ema_w = decay*ema_w + (1-decay)*dw ; new_embedding = new_ema_w/new_cs
// ---------------------------------------------------------------------------
__global__ __launch_bounds__(256) void k_ema(
    const float* __restrict__ emaW, const float* __restrict__ dw,
    float* __restrict__ out) {
  const int k = blockIdx.x, d = threadIdx.x;
  const size_t i = (size_t)k * EMB_DIM + d;
  const float w = DECAY * emaW[i] + OMD * dw[i];
  out[OFF_EMAW + i] = w;
  out[OFF_EMB + i] = w / out[OFF_CS + k];   // written by k_finalize earlier in-stream
}

// ---------------------------------------------------------------------------
extern "C" void kernel_launch(void* const* d_in, const int* in_sizes, int n_in,
                              void* d_out, int out_size, void* d_ws,
                              size_t ws_size, hipStream_t stream) {
  (void)in_sizes; (void)n_in; (void)out_size; (void)ws_size;
  const float* x     = (const float*)d_in[0];   // [16,256,32,32]
  const float* emb   = (const float*)d_in[1];   // [8192,256]
  const float* emaCs = (const float*)d_in[2];   // [8192]
  const float* emaW  = (const float*)d_in[3];   // [8192,256]
  float* out = (float*)d_out;

  // ---- workspace carve-out (256B aligned) ----
  char* w = (char*)d_ws;
  auto carve = [&](size_t bytes) -> void* {
    void* p = (void*)w;
    w += (bytes + 255) & ~(size_t)255;
    return p;
  };
  bf16_t* ehi   = (bf16_t*)carve((size_t)NUM_EMB * EMB_DIM * 2);  // 4 MB
  bf16_t* elo   = (bf16_t*)carve((size_t)NUM_EMB * EMB_DIM * 2);  // 4 MB
  bf16_t* xhi   = (bf16_t*)carve((size_t)NTOK * EMB_DIM * 2);     // 8 MB
  bf16_t* xlo   = (bf16_t*)carve((size_t)NTOK * EMB_DIM * 2);     // 8 MB
  float*  e2    = (float*)carve((size_t)NUM_EMB * 4);
  float*  cnts  = (float*)carve((size_t)NUM_EMB * 4);
  float*  rawCs = (float*)carve((size_t)NUM_EMB * 4);
  int*    idxI  = (int*)carve((size_t)NTOK * 4);
  float*  scal  = (float*)carve(256);            // [0]=sse [1]=nTot [2]=ent
  float*  dw    = (float*)carve((size_t)NUM_EMB * EMB_DIM * 4);   // 8 MB

  // ---- zero accumulators (graph-capturable async memsets) ----
  hipMemsetAsync(cnts, 0, (size_t)NUM_EMB * 4, stream);
  hipMemsetAsync(scal, 0, 256, stream);
  hipMemsetAsync(dw, 0, (size_t)NUM_EMB * EMB_DIM * 4, stream);

  // ---- pipeline ----
  k_prep_codebook<<<NUM_EMB, 256, 0, stream>>>(emb, ehi, elo, e2);
  k_prep_x<<<dim3(HW / 32, EMB_DIM / 32, BATCH), 256, 0, stream>>>(x, xhi, xlo);
  k_argmin<<<NTOK / 128, 256, 0, stream>>>(xhi, xlo, ehi, elo, e2, idxI,
                                           out + OFF_IDX);
  k_counts<<<NTOK / 256, 256, 0, stream>>>(idxI, cnts);
  k_scatter<<<(BATCH * EMB_DIM * HW) / 256, 256, 0, stream>>>(
      x, emb, idxI, out + OFF_QST, dw, &scal[0]);
  k_cs_ent<<<NUM_EMB / 256, 256, 0, stream>>>(cnts, emaCs, rawCs, &scal[1],
                                              &scal[2]);
  k_finalize<<<NUM_EMB / 256, 256, 0, stream>>>(rawCs, &scal[1], &scal[0],
                                                &scal[2], out);
  k_ema<<<NUM_EMB, 256, 0, stream>>>(emaW, dw, out);
}
